// SpectralWavelet_47012712022483
// MI455X (gfx1250) — compile-verified, matrix operand order checked by reading, not run on hardware
//
#include <hip/hip_runtime.h>

// ---------------- problem constants ----------------
constexpr int B     = 8;
constexpr int S     = 2048;
constexpr int DIN   = 2048;
constexpr int DOUT  = 2048;
constexpr int DC    = 64;
constexpr int RANK  = 12;

// ---------------- CDNA5 WMMA types ----------------
typedef __attribute__((ext_vector_type(16))) __bf16 v16bf;
typedef __attribute__((ext_vector_type(8)))  float  v8f;
typedef int v4i __attribute__((vector_size(16)));   // matches async builtin's V4i pointee

#define GLOBAL_AS __attribute__((address_space(1)))
#define LDS_AS    __attribute__((address_space(3)))

// gfx1250 async global->LDS copy builtins (codegen-confirmed in round 2).
#if defined(__has_builtin)
# if __has_builtin(__builtin_amdgcn_global_load_async_to_lds_b128) && \
     __has_builtin(__builtin_amdgcn_s_wait_asynccnt)
#  define ASYNC_LDS 1
# endif
#endif
#ifndef ASYNC_LDS
# define ASYNC_LDS 0
#endif

__device__ __forceinline__ unsigned short f2bf(float f) {
    union { float f; unsigned u; } v; v.f = f;
    unsigned r = v.u + 0x7fffu + ((v.u >> 16) & 1u);   // round-to-nearest-even
    return (unsigned short)(r >> 16);
}

// ---------------- kernel 0: zero small accumulators ----------------
__global__ void k_zero(float* gacc) {
    if (threadIdx.x < B) gacc[threadIdx.x] = 0.0f;
}

// ---------------- kernel 1: x -> bf16, fused gate partial sums ----------------
__global__ __launch_bounds__(256) void k_prep_x(const float* __restrict__ x,
                                                const float* __restrict__ gw,
                                                unsigned short* __restrict__ xb,
                                                float* __restrict__ gacc) {
    const int row = blockIdx.x;            // 0 .. B*S-1
    const int b   = row >> 11;             // S == 2048
    const float*     xr = x  + (size_t)row * DIN;
    unsigned short*  xo = xb + (size_t)row * DIN;
    float sum = 0.0f;
    for (int i = threadIdx.x; i < DIN; i += 256) {
        float v = xr[i];
        xo[i]   = f2bf(v);
        sum    += v * gw[i];
    }
    __shared__ float red[256];
    red[threadIdx.x] = sum;
    __syncthreads();
    for (int s = 128; s > 0; s >>= 1) {
        if (threadIdx.x < s) red[threadIdx.x] += red[threadIdx.x + s];
        __syncthreads();
    }
    if (threadIdx.x == 0) atomicAdd(&gacc[b], red[0]);
}

// ---------------- kernel 2: per-batch scalars ----------------
__global__ void k_scalars(const float* __restrict__ coords,
                          const float* __restrict__ cdir,
                          const float* __restrict__ gacc,
                          const float* __restrict__ gate_b,
                          float* __restrict__ ccos,
                          float* __restrict__ csin) {
    const int b = threadIdx.x;
    if (b < B) {
        float p = 0.0f;
        for (int i = 0; i < DC; ++i) p += coords[b * DC + i] * cdir[i];
        float z    = gacc[b] / (float)S + gate_b[0];
        float gate = 1.0f / (1.0f + expf(-z));
        ccos[b] = gate * cosf(p);
        csin[b] = gate * sinf(p);
    }
}

// ---------------- kernel 3: W_eff(b) = W_base + rank-24 update, in bf16 ----------------
__global__ __launch_bounds__(256) void k_weff(const float* __restrict__ Wb,
                                              const float* __restrict__ Uc,
                                              const float* __restrict__ Vc,
                                              const float* __restrict__ Us,
                                              const float* __restrict__ Vs,
                                              const float* __restrict__ ccos,
                                              const float* __restrict__ csin,
                                              unsigned short* __restrict__ weff) {
    const size_t gidx = (size_t)blockIdx.x * 256 + threadIdx.x;   // B*DOUT*DIN threads
    const int b = (int)(gidx >> 22);                              // DOUT*DIN == 1<<22
    const int r = (int)(gidx & ((1u << 22) - 1));
    const int o = r >> 11;                                        // DIN == 2048
    const int d = r & (DIN - 1);
    float dc = 0.0f, ds = 0.0f;
#pragma unroll
    for (int k = 0; k < RANK; ++k) {
        dc += Uc[o * RANK + k] * Vc[d * RANK + k];
        ds += Us[o * RANK + k] * Vs[d * RANK + k];
    }
    float v = Wb[(size_t)o * DIN + d] + ccos[b] * dc + csin[b] * ds;
    weff[gidx] = f2bf(v);
}

// ---------------- kernel 4: batched bf16 WMMA GEMM, async double-buffered ----------------
// C[b, s, o] = sum_d xb[b,s,d] * weff[b,o,d]  (f32 accumulate)
// block = 256 threads = 8 waves; block tile 128(M=s) x 256(N=o); K step 32.
// wave grid 2(M) x 4(N); each wave: 64x64 = 4x4 frags of 16x16 -> 16 WMMA / K-step.
union Frag { v16bf v; uint4 q[2]; };

constexpr int KT      = 32;            // K elements per step
constexpr int NSTEPS  = DIN / KT;      // 64
constexpr int ACH     = (128 * KT) / 8 / 256;  // A 16B-chunks per thread = 2
constexpr int BCH     = (256 * KT) / 8 / 256;  // B 16B-chunks per thread = 4

__global__ __launch_bounds__(256) void k_gemm(const unsigned short* __restrict__ xb,
                                              const unsigned short* __restrict__ wb,
                                              float* __restrict__ out) {
    const int b  = blockIdx.z;
    const int o0 = blockIdx.x * 256;
    const int s0 = blockIdx.y * 128;

    __shared__ __attribute__((aligned(16))) unsigned short sA[2][128 * KT];  // 2 x 8 KB
    __shared__ __attribute__((aligned(16))) unsigned short sB[2][256 * KT];  // 2 x 16 KB

    const int t    = threadIdx.x;
    const int lane = t & 31;
    const int w    = t >> 5;
    const int wm   = w & 1;        // M offset wm*64
    const int wn   = w >> 1;       // N offset wn*64
    const int g    = lane >> 4;    // K-half select (16-bit A/B layout)
    const int lm   = lane & 15;    // row within 16x16 tile

    const v8f vzero = {0.f, 0.f, 0.f, 0.f, 0.f, 0.f, 0.f, 0.f};
    v8f acc[4][4];
#pragma unroll
    for (int mi = 0; mi < 4; ++mi)
#pragma unroll
        for (int ni = 0; ni < 4; ++ni) acc[mi][ni] = vzero;

    const unsigned short* Abase = xb + ((size_t)b * S    + s0) * DIN;
    const unsigned short* Bbase = wb + ((size_t)b * DOUT + o0) * DIN;

    // issue one tile's async loads into buffer `buf` for K offset k0
    auto issue = [&](int buf, int k0) {
#pragma unroll
        for (int i = 0; i < ACH; ++i) {
            const int q   = t + i * 256;       // A: 512 chunks
            const int row = q >> 2, c = q & 3;
            const size_t goff = (size_t)row * DIN + k0 + c * 8;
#if ASYNC_LDS
            __builtin_amdgcn_global_load_async_to_lds_b128(
                (GLOBAL_AS v4i*)(Abase + goff),
                (LDS_AS v4i*)(&sA[buf][row * KT + c * 8]), 0, 0);
#else
            *(uint4*)(&sA[buf][row * KT + c * 8]) = *(const uint4*)(Abase + goff);
#endif
            if (k0 + 2 * KT < DIN)
                __builtin_prefetch(Abase + goff + 2 * KT, 0, 1);  // global_prefetch_b8, 2 tiles ahead
        }
#pragma unroll
        for (int i = 0; i < BCH; ++i) {
            const int q   = t + i * 256;       // B: 1024 chunks
            const int row = q >> 2, c = q & 3;
            const size_t goff = (size_t)row * DIN + k0 + c * 8;
#if ASYNC_LDS
            __builtin_amdgcn_global_load_async_to_lds_b128(
                (GLOBAL_AS v4i*)(Bbase + goff),
                (LDS_AS v4i*)(&sB[buf][row * KT + c * 8]), 0, 0);
#else
            *(uint4*)(&sB[buf][row * KT + c * 8]) = *(const uint4*)(Bbase + goff);
#endif
            if (k0 + 2 * KT < DIN)
                __builtin_prefetch(Bbase + goff + 2 * KT, 0, 1);
        }
    };

    issue(0, 0);   // prologue: fill buffer 0

    for (int kt = 0; kt < NSTEPS; ++kt) {
        const int cur = kt & 1;
        if (kt + 1 < NSTEPS) {
            issue(cur ^ 1, (kt + 1) * KT);      // overlap next tile with this compute
#if ASYNC_LDS
            // 6 loads/thread just issued; wait until only those remain -> previous tile landed
            __builtin_amdgcn_s_wait_asynccnt(ACH + BCH);
#endif
        } else {
#if ASYNC_LDS
            __builtin_amdgcn_s_wait_asynccnt(0);
#endif
        }
        __syncthreads();   // tile `cur` visible to all waves

        Frag fa[4], fb[4];
#pragma unroll
        for (int mi = 0; mi < 4; ++mi) {
            const int row = wm * 64 + mi * 16 + lm;
            fa[mi].q[0] = *(const uint4*)(&sA[cur][row * KT + g * 8]);
            fa[mi].q[1] = *(const uint4*)(&sA[cur][row * KT + 16 + g * 8]);
        }
#pragma unroll
        for (int ni = 0; ni < 4; ++ni) {
            const int row = wn * 64 + ni * 16 + lm;
            fb[ni].q[0] = *(const uint4*)(&sB[cur][row * KT + g * 8]);
            fb[ni].q[1] = *(const uint4*)(&sB[cur][row * KT + 16 + g * 8]);
        }
#pragma unroll
        for (int mi = 0; mi < 4; ++mi)
#pragma unroll
            for (int ni = 0; ni < 4; ++ni)
                acc[mi][ni] = __builtin_amdgcn_wmma_f32_16x16x32_bf16(
                    false, fa[mi].v, false, fb[ni].v,
                    (short)0, acc[mi][ni], false, false);

        __syncthreads();   // all waves done reading `cur` before it is refilled
    }

    // Epilogue: C/D layout — lane L, vgpr j -> M = (L/16)*8 + j, N = L%16
#pragma unroll
    for (int mi = 0; mi < 4; ++mi)
#pragma unroll
        for (int ni = 0; ni < 4; ++ni) {
            const int srow = s0 + wm * 64 + mi * 16 + g * 8;
            const int col  = o0 + wn * 64 + ni * 16 + lm;
            float* po = out + ((size_t)b * S + srow) * DOUT + col;
#pragma unroll
            for (int j = 0; j < 8; ++j) po[(size_t)j * DOUT] = acc[mi][ni][j];
        }
}

// ---------------- host launcher ----------------
extern "C" void kernel_launch(void* const* d_in, const int* in_sizes, int n_in,
                              void* d_out, int out_size, void* d_ws, size_t ws_size,
                              hipStream_t stream) {
    const float* x       = (const float*)d_in[0];
    const float* coords  = (const float*)d_in[1];
    const float* W_base  = (const float*)d_in[2];
    const float* cdir    = (const float*)d_in[3];
    const float* U_cos   = (const float*)d_in[4];
    const float* V_cos   = (const float*)d_in[5];
    const float* U_sin   = (const float*)d_in[6];
    const float* V_sin   = (const float*)d_in[7];
    const float* gate_w  = (const float*)d_in[8];
    const float* gate_b  = (const float*)d_in[9];
    float* out = (float*)d_out;

    char* ws = (char*)d_ws;
    const size_t XB_BYTES = (size_t)B * S    * DIN * sizeof(unsigned short); // 64 MiB
    const size_t WE_BYTES = (size_t)B * DOUT * DIN * sizeof(unsigned short); // 64 MiB
    unsigned short* xb   = (unsigned short*)ws;
    unsigned short* weff = (unsigned short*)(ws + XB_BYTES);
    float* gacc = (float*)(ws + XB_BYTES + WE_BYTES);
    float* ccos = gacc + B;
    float* csin = ccos + B;

    k_zero<<<1, 32, 0, stream>>>(gacc);
    k_prep_x<<<B * S, 256, 0, stream>>>(x, gate_w, xb, gacc);
    k_scalars<<<1, 64, 0, stream>>>(coords, cdir, gacc, gate_b, ccos, csin);
    k_weff<<<(int)(((size_t)B * DOUT * DIN) / 256), 256, 0, stream>>>(
        W_base, U_cos, V_cos, U_sin, V_sin, ccos, csin, weff);

    dim3 grid(DOUT / 256, S / 128, B);
    k_gemm<<<grid, 256, 0, stream>>>(xb, weff, out);
}
